// GRU_78632261255390
// MI455X (gfx1250) — matrix operand order repaired and checked by reading, not compiled, not run on hardware
//
#include <hip/hip_runtime.h>
#include <hip/hip_bf16.h>

// ---------------------------------------------------------------------------
// GRU (2 layers) + FC for MI455X (gfx1250), bf16 WMMA with f32 accumulation.
//
// Shapes: I=256, H=512, O=256, B=64, T=1024.
// Strategy:
//   * Convert all weights and the (transposed to time-major) input to bf16.
//   * Per timestep: two kernels (layer0, layer1). Each kernel = 8 WGs x 512
//     threads; each wave computes one 16x16 (batch x hidden) tile, holding
//     six f32 accumulators (gx_r/z/n, gh_r/z/n), K-looped with
//     v_wmma_f32_16x16x32_bf16, then applies the fused GRU gate math.
//     h state is ping-ponged so no WG reads what another WG writes.
//   * Layer-1 h is also written (bf16) to an [T*B, H] buffer; one big WMMA
//     GEMM computes the final FC into d_out (f32).
// ---------------------------------------------------------------------------

typedef unsigned short u16;
typedef __attribute__((ext_vector_type(16))) __bf16 v16bf;
typedef __attribute__((ext_vector_type(8)))  float  v8f;

#define GI 256
#define GH 512
#define GO 256
#define GB 64
#define GT 1024

union FragB16 {
    v16bf v;
    uint4 q[2];
};

__device__ __forceinline__ u16 f2bf(float f) {
    union { float f; unsigned u; } v;
    v.f = f;
    unsigned r = v.u + 0x7FFFu + ((v.u >> 16) & 1u);  // RNE
    return (u16)(r >> 16);
}

__device__ __forceinline__ float bf2f(u16 b) {
    union { unsigned u; float f; } v;
    v.u = ((unsigned)b) << 16;
    return v.f;
}

// A fragment: 16x32 bf16 tile of row-major [*, lda] matrix, rows m0..m0+15,
// K columns kglob..kglob+31.  ISA layout: lane(0..15) M=lane, holds
// K = kb+0..7 (elems 0..7) and K = 16+kb+0..7 (elems 8..15), kb = (lane>=16)*8.
__device__ __forceinline__ v16bf load_frag_A(const u16* __restrict__ base,
                                             int lda, int m0, int kglob, int lane) {
    int row = m0 + (lane & 15);
    int kb  = (lane >> 4) * 8;
    const u16* p = base + (size_t)row * lda + kglob + kb;
    FragB16 f;
    f.q[0] = *(const uint4*)(p);        // K = kglob+kb   .. +7
    f.q[1] = *(const uint4*)(p + 16);   // K = kglob+16+kb .. +7
    return f.v;
}

// B fragment: 32x16 bf16 tile of B = W^T, where W is row-major [*, ldb].
// B column n == W row (n0 + lane&15); lane holds K = kglob+kb+0..7 (elems 0..7)
// and K = kglob+16+kb+0..7 (elems 8..15), kb = (lane>=16)*8.
__device__ __forceinline__ v16bf load_frag_B(const u16* __restrict__ wbase,
                                             int ldb, int n0, int kglob, int lane) {
    int row = n0 + (lane & 15);
    int kb  = (lane >> 4) * 8;
    const u16* p = wbase + (size_t)row * ldb + kglob + kb;
    FragB16 f;
    f.q[0] = *(const uint4*)(p);
    f.q[1] = *(const uint4*)(p + 16);
    return f.v;
}

#define WMMA_BF16(A, Bf, C) \
    __builtin_amdgcn_wmma_f32_16x16x32_bf16(false, (A), false, (Bf), (short)0, (C), false, false)

// ---------------------------------------------------------------------------
// One GRU layer step.  grid = H/64 WGs, block = 512 (16 waves).
// Wave w -> batch tile mT = (w&3)*16, hidden tile nT = (w>>2)*16 within the
// WG's 64-column slice.  x: [B, Kx] bf16.  Wih: [3H, Kx] bf16 row-major.
// Whh: [3H, H] bf16 row-major.  hprev/hnext: [B, H] bf16 (ping-pong).
// ---------------------------------------------------------------------------
__global__ __launch_bounds__(512) void gru_step_kernel(
    const u16* __restrict__ x, int Kx,
    const u16* __restrict__ Wih,
    const u16* __restrict__ Whh,
    const float* __restrict__ bih,
    const float* __restrict__ bhh,
    const u16* __restrict__ hprev,
    u16* __restrict__ hnext,
    u16* __restrict__ out2)  // layer-1: also write to outs row block; else null
{
    const int lane = threadIdx.x & 31;
    const int wave = threadIdx.x >> 5;
    const int mT = (wave & 3) << 4;                 // batch tile base (0..48)
    const int j0 = blockIdx.x * 64 + ((wave >> 2) << 4);  // hidden column base

    v8f ax_r = {}, ax_z = {}, ax_n = {};
    v8f ah_r = {}, ah_z = {}, ah_n = {};

    // gx = x @ Wih^T   (three gate slices)
    for (int k = 0; k < Kx; k += 32) {
        v16bf a  = load_frag_A(x, Kx, mT, k, lane);
        v16bf br = load_frag_B(Wih, Kx, 0 * GH + j0, k, lane);
        v16bf bz = load_frag_B(Wih, Kx, 1 * GH + j0, k, lane);
        v16bf bn = load_frag_B(Wih, Kx, 2 * GH + j0, k, lane);
        ax_r = WMMA_BF16(a, br, ax_r);
        ax_z = WMMA_BF16(a, bz, ax_z);
        ax_n = WMMA_BF16(a, bn, ax_n);
    }
    // gh = h_prev @ Whh^T
    for (int k = 0; k < GH; k += 32) {
        v16bf a  = load_frag_A(hprev, GH, mT, k, lane);
        v16bf br = load_frag_B(Whh, GH, 0 * GH + j0, k, lane);
        v16bf bz = load_frag_B(Whh, GH, 1 * GH + j0, k, lane);
        v16bf bn = load_frag_B(Whh, GH, 2 * GH + j0, k, lane);
        ah_r = WMMA_BF16(a, br, ah_r);
        ah_z = WMMA_BF16(a, bz, ah_z);
        ah_n = WMMA_BF16(a, bn, ah_n);
    }

    // Fused GRU gates.  Lane owns column j, rows mT+rowoff+r (r=0..7).
    const int col    = lane & 15;
    const int rowoff = (lane >> 4) * 8;
    const int j      = j0 + col;
    const float bir = bih[j],          bhr = bhh[j];
    const float biz = bih[GH + j],     bhz = bhh[GH + j];
    const float bin = bih[2 * GH + j], bhn = bhh[2 * GH + j];

#pragma unroll
    for (int r = 0; r < 8; ++r) {
        const int b = mT + rowoff + r;
        float gr = ax_r[r] + bir + ah_r[r] + bhr;
        float gz = ax_z[r] + biz + ah_z[r] + bhz;
        float rg = 1.0f / (1.0f + __expf(-gr));
        float zg = 1.0f / (1.0f + __expf(-gz));
        float nn = tanhf(ax_n[r] + bin + rg * (ah_n[r] + bhn));
        float hp = bf2f(hprev[(size_t)b * GH + j]);
        float hv = nn + zg * (hp - nn);
        u16 hb = f2bf(hv);
        hnext[(size_t)b * GH + j] = hb;
        if (out2) out2[(size_t)b * GH + j] = hb;
    }
}

// ---------------------------------------------------------------------------
// Final FC: out[M, 256] = A[M, 512] @ Wfc[256, 512]^T + b.  M = T*B = 65536.
// grid = (M/64, 256/64), block = 512 (16 waves, one 16x16 tile each).
// ---------------------------------------------------------------------------
__global__ __launch_bounds__(512) void fc_kernel(
    const u16* __restrict__ A,
    const u16* __restrict__ Wfc,
    const float* __restrict__ bfc,
    float* __restrict__ out)
{
    const int lane = threadIdx.x & 31;
    const int wave = threadIdx.x >> 5;
    const int mT = blockIdx.x * 64 + ((wave & 3) << 4);
    const int nT = blockIdx.y * 64 + ((wave >> 2) << 4);

    v8f acc = {};
    for (int k = 0; k < GH; k += 32) {
        v16bf a = load_frag_A(A,   GH, mT, k, lane);
        v16bf b = load_frag_B(Wfc, GH, nT, k, lane);
        acc = WMMA_BF16(a, b, acc);
    }
    const int col = nT + (lane & 15);
    const int row = mT + (lane >> 4) * 8;
    const float bias = bfc[col];
#pragma unroll
    for (int r = 0; r < 8; ++r)
        out[(size_t)(row + r) * GO + col] = acc[r] + bias;
}

// ---------------------------------------------------------------------------
// Conversion / init helpers
// ---------------------------------------------------------------------------
__global__ void cvt_f32_bf16_kernel(const float* __restrict__ in,
                                    u16* __restrict__ out, int n) {
    for (int i = blockIdx.x * blockDim.x + threadIdx.x; i < n;
         i += gridDim.x * blockDim.x)
        out[i] = f2bf(in[i]);
}

// input [B, T, I] f32  ->  xbf [T, B, I] bf16 (time-major for per-step tiles)
__global__ void cvt_input_kernel(const float* __restrict__ in,
                                 u16* __restrict__ out) {
    int idx = blockIdx.x * blockDim.x + threadIdx.x;
    if (idx >= GT * GB * GI) return;
    int i = idx % GI;
    int r = idx / GI;
    int b = r % GB;
    int t = r / GB;
    out[idx] = f2bf(in[((size_t)b * GT + t) * GI + i]);
}

__global__ void zero_bf16_kernel(u16* __restrict__ p, int n) {
    for (int i = blockIdx.x * blockDim.x + threadIdx.x; i < n;
         i += gridDim.x * blockDim.x)
        p[i] = 0;
}

// ---------------------------------------------------------------------------
// Host-side orchestration (graph-capture safe: only kernel launches).
// Workspace layout (u16 elements), total ~107 MB:
//   wih0 [3H*I] | whh0 [3H*H] | wih1 [3H*H] | whh1 [3H*H] | wfc [O*H]
//   | xbf [T*B*I] | outs [T*B*H] | h0 ping-pong [2*B*H] | h1 ping-pong [2*B*H]
// ---------------------------------------------------------------------------
extern "C" void kernel_launch(void* const* d_in, const int* in_sizes, int n_in,
                              void* d_out, int out_size, void* d_ws, size_t ws_size,
                              hipStream_t stream) {
    const float* input = (const float*)d_in[0];
    const float* W_ih0 = (const float*)d_in[1];
    const float* b_ih0 = (const float*)d_in[2];
    const float* W_hh0 = (const float*)d_in[3];
    const float* b_hh0 = (const float*)d_in[4];
    const float* W_ih1 = (const float*)d_in[5];
    const float* b_ih1 = (const float*)d_in[6];
    const float* W_hh1 = (const float*)d_in[7];
    const float* b_hh1 = (const float*)d_in[8];
    const float* W_fc  = (const float*)d_in[9];
    const float* b_fc  = (const float*)d_in[10];

    u16* ws = (u16*)d_ws;
    size_t off = 0;
    u16* wih0 = ws + off; off += (size_t)3 * GH * GI;   // 393216
    u16* whh0 = ws + off; off += (size_t)3 * GH * GH;   // 786432
    u16* wih1 = ws + off; off += (size_t)3 * GH * GH;
    u16* whh1 = ws + off; off += (size_t)3 * GH * GH;
    u16* wfc  = ws + off; off += (size_t)GO * GH;       // 131072
    u16* xbf  = ws + off; off += (size_t)GT * GB * GI;  // 16.7M
    u16* outs = ws + off; off += (size_t)GT * GB * GH;  // 33.5M
    u16* h0   = ws + off; off += (size_t)2 * GB * GH;   // ping-pong
    u16* h1   = ws + off; off += (size_t)2 * GB * GH;

    const int cvtB = 256;
    auto blocks = [](int n, int b) { return (n + b - 1) / b; };

    // Convert weights to bf16 (L2-resident thereafter).
    cvt_f32_bf16_kernel<<<blocks(3 * GH * GI, cvtB), cvtB, 0, stream>>>(W_ih0, wih0, 3 * GH * GI);
    cvt_f32_bf16_kernel<<<blocks(3 * GH * GH, cvtB), cvtB, 0, stream>>>(W_hh0, whh0, 3 * GH * GH);
    cvt_f32_bf16_kernel<<<blocks(3 * GH * GH, cvtB), cvtB, 0, stream>>>(W_ih1, wih1, 3 * GH * GH);
    cvt_f32_bf16_kernel<<<blocks(3 * GH * GH, cvtB), cvtB, 0, stream>>>(W_hh1, whh1, 3 * GH * GH);
    cvt_f32_bf16_kernel<<<blocks(GO * GH, cvtB), cvtB, 0, stream>>>(W_fc, wfc, GO * GH);

    // Transpose input to time-major bf16.
    cvt_input_kernel<<<blocks(GT * GB * GI, cvtB), cvtB, 0, stream>>>(input, xbf);

    // Zero initial hidden states (both ping-pong slots for determinism).
    zero_bf16_kernel<<<blocks(2 * GB * GH, cvtB), cvtB, 0, stream>>>(h0, 2 * GB * GH);
    zero_bf16_kernel<<<blocks(2 * GB * GH, cvtB), cvtB, 0, stream>>>(h1, 2 * GB * GH);

    // Sequential recurrence: 2 kernels per timestep, ping-pong h buffers.
    const int slot = GB * GH;
    for (int t = 0; t < GT; ++t) {
        const int p = t & 1;
        gru_step_kernel<<<GH / 64, 512, 0, stream>>>(
            xbf + (size_t)t * GB * GI, GI,
            wih0, whh0, b_ih0, b_hh0,
            h0 + p * slot, h0 + (1 - p) * slot, (u16*)nullptr);
        gru_step_kernel<<<GH / 64, 512, 0, stream>>>(
            h0 + (1 - p) * slot, GH,
            wih1, whh1, b_ih1, b_hh1,
            h1 + p * slot, h1 + (1 - p) * slot,
            outs + (size_t)t * GB * GH);
    }

    // Final FC over all T*B rows.
    dim3 fgrid((GT * GB) / 64, GO / 64);
    fc_kernel<<<fgrid, 512, 0, stream>>>(outs, wfc, b_fc, (float*)d_out);
}
